// ProtoTrexW_73280732004828
// MI455X (gfx1250) — compile-verified, matrix operand order checked by reading, not run on hardware
//
#include <hip/hip_runtime.h>
#include <hip/hip_bf16.h>

// Problem constants (from reference)
#define NN 16
#define SS 128
#define EE 1024
#define PP 128
#define CCL 5
#define KE 4096          // K*E
#define EPSV 1e-8f

// Workspace layout
// floats:
#define XN2_OFF 0        // 2*16*128 row-norm^2 (padded stride 128)
#define PN2_OFF 4096     // 128 prototype-norm^2
#define PART_OFF 4224    // 2*16*64*8 per-htile partial mins
#define SPLIT_OFF 20608  // float offset where bf16 arrays begin (16B aligned)
// then (ushort): proto_hi[128*4096], proto_lo[same],
//                X0_hi[16*125*4096], X0_lo[same],  (branch d=1 gathered stream)
//                X1_hi[16*122*4096], X1_lo[same]   (branch d=2 gathered stream)
// total ws ~ 0.1 MB + 67 MB (fits comfortably in the 192 MB L2)

typedef __attribute__((ext_vector_type(16))) __bf16 v16bf;
typedef __attribute__((ext_vector_type(8)))  float  v8f;
typedef __attribute__((ext_vector_type(8)))  unsigned short us8;
typedef __attribute__((ext_vector_type(4)))  unsigned short us4;

union BFVec { v16bf v; us8 h[2]; };

// ---- fp32 -> bf16 split (hi + lo), round-to-nearest-even ----
__device__ __forceinline__ unsigned short f2bf(float f) {
  unsigned u = __float_as_uint(f);
  unsigned r = u + 0x7FFFu + ((u >> 16) & 1u);
  return (unsigned short)(r >> 16);
}
__device__ __forceinline__ float bf2f(unsigned short h) {
  return __uint_as_float(((unsigned)h) << 16);
}
__device__ __forceinline__ void split4(float4 x, us4* h4, us4* l4) {
  us4 h, l;
  unsigned short t;
  t = f2bf(x.x); h[0] = t; l[0] = f2bf(x.x - bf2f(t));
  t = f2bf(x.y); h[1] = t; l[1] = f2bf(x.y - bf2f(t));
  t = f2bf(x.z); h[2] = t; l[2] = f2bf(x.z - bf2f(t));
  t = f2bf(x.w); h[3] = t; l[3] = f2bf(x.w - bf2f(t));
  *h4 = h; *l4 = l;
}

// xor-swizzle wave32 sum (ds_swizzle group-of-32: offset = xor<<10 | 0x1f)
__device__ __forceinline__ float wave_sum32(float v) {
  v += __int_as_float(__builtin_amdgcn_ds_swizzle(__float_as_int(v), 0x041F));
  v += __int_as_float(__builtin_amdgcn_ds_swizzle(__float_as_int(v), 0x081F));
  v += __int_as_float(__builtin_amdgcn_ds_swizzle(__float_as_int(v), 0x101F));
  v += __int_as_float(__builtin_amdgcn_ds_swizzle(__float_as_int(v), 0x201F));
  v += __int_as_float(__builtin_amdgcn_ds_swizzle(__float_as_int(v), 0x401F));
  return v;
}

// -------- Kernel 0a: fp32 -> (bf16_hi, bf16_lo) split (contiguous) --------
__global__ __launch_bounds__(256) void split_kernel(const float* __restrict__ src,
                                                    unsigned short* __restrict__ dh,
                                                    unsigned short* __restrict__ dl,
                                                    int nquad) {
  const int i = blockIdx.x * 256 + threadIdx.x;
  if (i >= nquad) return;
  const float4 x = ((const float4*)src)[i];
  split4(x, (us4*)(dh + 4 * (size_t)i), (us4*)(dl + 4 * (size_t)i));
}

// -------- Kernel 0b: materialize gathered stream, split to bf16 hi/lo ------
// U_flat[n] = concat_k emb[n].flat[D*k*E : D*k*E + H*E]; rows of the (H,4096)
// A matrix are contiguous slices of this stream, so GEMM A-loads become linear.
template<int D, int H>
__global__ __launch_bounds__(256) void gather_split_kernel(
    const float* __restrict__ emb,
    unsigned short* __restrict__ xh, unsigned short* __restrict__ xl) {
  constexpr int LEN = H * KE;            // stream elements per n
  constexpr int HEc = H * EE;
  const int q = blockIdx.x * 256 + threadIdx.x;   // float4 index
  const int e0 = q * 4;
  if (e0 >= NN * LEN) return;
  const int n = e0 / LEN;                // constexpr divisors -> mul/shift
  const int g = e0 - n * LEN;
  const int k = g / HEc;                 // 4-aligned quads never straddle HEc
  const float4 x = *(const float4*)(emb + (size_t)n * SS * EE +
                                    (size_t)(D * k) * EE + (g - k * HEc));
  const size_t o = (size_t)n * LEN + g;
  split4(x, (us4*)(xh + o), (us4*)(xl + o));
}

// -------- Kernel 1: row norms (gathered X rows) + prototype norms --------
__global__ __launch_bounds__(32) void norms_kernel(const float* __restrict__ emb,
                                                   const float* __restrict__ proto,
                                                   float* __restrict__ ws) {
  const int task = blockIdx.x;
  const int lane = threadIdx.x;
  float acc = 0.f;
  if (task < 2 * NN * SS) {
    const int b  = task >> 11;
    const int n  = (task >> 7) & (NN - 1);
    const int h2 = task & (SS - 1);
    const int H  = b ? 122 : 125;
    const int d  = b + 1;
    if (h2 >= H) return;                 // uniform across the wave
    const int HE = H * EE;
    const float* eb = emb + (size_t)n * SS * EE;
    const int gbase = h2 * KE + lane * 4;
    for (int it = 0; it < 32; ++it) {
      int g = gbase + it * 128;
      int k = g / HE;
      const float4 v = *(const float4*)(eb + (size_t)(d * k) * EE + (g - k * HE));
      acc = fmaf(v.x, v.x, acc);
      acc = fmaf(v.y, v.y, acc);
      acc = fmaf(v.z, v.z, acc);
      acc = fmaf(v.w, v.w, acc);
    }
    acc = wave_sum32(acc);
    if (lane == 0) ws[XN2_OFF + task] = acc;
  } else {
    const int p = task - 2 * NN * SS;
    const float* pr = proto + (size_t)p * KE;
    for (int it = 0; it < 32; ++it) {
      const float4 v = *(const float4*)(pr + lane * 4 + it * 128);
      acc = fmaf(v.x, v.x, acc);
      acc = fmaf(v.y, v.y, acc);
      acc = fmaf(v.z, v.z, acc);
      acc = fmaf(v.w, v.w, acc);
    }
    acc = wave_sum32(acc);
    if (lane == 0) ws[PN2_OFF + p] = acc;
  }
}

// -------- Kernel 2/3: WMMA GEMM, 16 rows x 32 protos per wave --------
// bf16x3 split WMMA; all operand loads are linear (pre-gathered streams), so
// the hot loop is 12x global_load_b128 + 6x v_wmma per 32 K-elements.
// A 16-bit WMMA layout: lane L holds M = L&15; element i holds
// K = i + 8*(L>>4) + (i>=8 ? 8 : 0). B mirrored with N = L&15.
template<int H, int J, int OV, int B>
__global__ __launch_bounds__(32) void proto_gemm(
    const unsigned short* __restrict__ xh, const unsigned short* __restrict__ xl,
    const unsigned short* __restrict__ ph, const unsigned short* __restrict__ pl,
    const float* __restrict__ maskp, float* __restrict__ ws) {
  const int lane  = threadIdx.x;
  const int hi    = lane >> 4;
  const int m15   = lane & 15;
  const int pt    = blockIdx.x;          // 0..1 (32 protos each)
  const int htile = blockIdx.y;          // 0..7 (16 rows each)
  const int n     = blockIdx.z;          // 0..15
  const int h2base = htile * 16;
  int ar = h2base + m15; if (ar > H - 1) ar = H - 1;   // clamp (masked later)

  const unsigned short* xbh = xh + (size_t)n * (H * KE) + ar * KE + 8 * hi;
  const unsigned short* xbl = xl + (size_t)n * (H * KE) + ar * KE + 8 * hi;
  const int prow0 = J + pt * 32 + m15;
  const int prow1 = prow0 + 16;
  const unsigned short* pbh0 = ph + (size_t)prow0 * KE + 8 * hi;
  const unsigned short* pbl0 = pl + (size_t)prow0 * KE + 8 * hi;
  const unsigned short* pbh1 = ph + (size_t)prow1 * KE + 8 * hi;
  const unsigned short* pbl1 = pl + (size_t)prow1 * KE + 8 * hi;

  v8f acc0 = {0.f, 0.f, 0.f, 0.f, 0.f, 0.f, 0.f, 0.f};
  v8f acc1 = {0.f, 0.f, 0.f, 0.f, 0.f, 0.f, 0.f, 0.f};
  for (int f0 = 0; f0 < KE; f0 += 32) {
    BFVec ah, al, bh0, bl0, bh1, bl1;
    ah.h[0]  = *(const us8*)(xbh + f0);   ah.h[1]  = *(const us8*)(xbh + f0 + 16);
    al.h[0]  = *(const us8*)(xbl + f0);   al.h[1]  = *(const us8*)(xbl + f0 + 16);
    bh0.h[0] = *(const us8*)(pbh0 + f0);  bh0.h[1] = *(const us8*)(pbh0 + f0 + 16);
    bl0.h[0] = *(const us8*)(pbl0 + f0);  bl0.h[1] = *(const us8*)(pbl0 + f0 + 16);
    bh1.h[0] = *(const us8*)(pbh1 + f0);  bh1.h[1] = *(const us8*)(pbh1 + f0 + 16);
    bl1.h[0] = *(const us8*)(pbl1 + f0);  bl1.h[1] = *(const us8*)(pbl1 + f0 + 16);
    // fp32-accurate dot via bf16 split: hi*hi + hi*lo + lo*hi; interleave the
    // two independent accumulator chains to avoid WMMA->WMMA RAW bubbles.
    acc0 = __builtin_amdgcn_wmma_f32_16x16x32_bf16(false, ah.v, false, bh0.v, (short)0, acc0, false, false);
    acc1 = __builtin_amdgcn_wmma_f32_16x16x32_bf16(false, ah.v, false, bh1.v, (short)0, acc1, false, false);
    acc0 = __builtin_amdgcn_wmma_f32_16x16x32_bf16(false, ah.v, false, bl0.v, (short)0, acc0, false, false);
    acc1 = __builtin_amdgcn_wmma_f32_16x16x32_bf16(false, ah.v, false, bl1.v, (short)0, acc1, false, false);
    acc0 = __builtin_amdgcn_wmma_f32_16x16x32_bf16(false, al.v, false, bh0.v, (short)0, acc0, false, false);
    acc1 = __builtin_amdgcn_wmma_f32_16x16x32_bf16(false, al.v, false, bh1.v, (short)0, acc1, false, false);
  }

  const float pnv0 = fmaxf(sqrtf(ws[PN2_OFF + prow0]), EPSV);
  const float pnv1 = fmaxf(sqrtf(ws[PN2_OFF + prow1]), EPSV);
  float dmin0 = __builtin_inff();
  float dmin1 = __builtin_inff();
#pragma unroll
  for (int v = 0; v < 8; ++v) {
    const int M  = v + 8 * hi;           // C/D layout: M = vgpr + 8*(lane>>4)
    const int h2 = h2base + M;
    const int hr = (h2 > H - 1) ? (H - 1) : h2;
    const float xn = fmaxf(sqrtf(ws[XN2_OFF + B * (NN * SS) + n * SS + hr]), EPSV);
    const float mv = (h2 == 0) ? 1.0f : maskp[n * SS + OV + hr];
    const float s  = mv / xn;            // shared row scale
    float d0 = -(acc0[v] / pnv0) * s;
    float d1 = -(acc1[v] / pnv1) * s;
    if (h2 >= H) { d0 = __builtin_inff(); d1 = __builtin_inff(); }
    dmin0 = fminf(dmin0, d0);
    dmin1 = fminf(dmin1, d1);
  }
  // combine lane L with lane L^16 (same proto column, other 8 rows)
  dmin0 = fminf(dmin0, __int_as_float(__builtin_amdgcn_ds_swizzle(__float_as_int(dmin0), 0x401F)));
  dmin1 = fminf(dmin1, __int_as_float(__builtin_amdgcn_ds_swizzle(__float_as_int(dmin1), 0x401F)));
  if (lane < 16) {
    float* basep = ws + PART_OFF + (size_t)((B * NN + n) * 64) * 8;
    basep[(pt * 32 + m15) * 8 + htile]      = dmin0;
    basep[(pt * 32 + 16 + m15) * 8 + htile] = dmin1;
  }
}

// -------- Kernel 4: min over htiles -> prototype_distances; then FC --------
__global__ __launch_bounds__(128) void finalize_kernel(const float* __restrict__ fcw,
                                                       const float* __restrict__ ws,
                                                       float* __restrict__ out) {
  __shared__ float sh[PP];
  const int n = blockIdx.x;
  const int p = threadIdx.x;
  const int b  = p >> 6;                 // branch 0: protos 0..63, branch 1: 64..127
  const int pl = p & 63;
  const float* base = ws + PART_OFF + ((size_t)((b * NN + n) * 64) + pl) * 8;
  float m = base[0];
#pragma unroll
  for (int t = 1; t < 8; ++t) m = fminf(m, base[t]);
  out[n * PP + p] = m;                   // prototype_distances
  sh[p] = m;
  __syncthreads();
  if (p < CCL) {
    float s = 0.f;
    for (int q = 0; q < PP; ++q) s = fmaf(sh[q], fcw[p * PP + q], s);
    out[NN * PP + n * CCL + p] = s;      // class_out
  }
}

extern "C" void kernel_launch(void* const* d_in, const int* in_sizes, int n_in,
                              void* d_out, int out_size, void* d_ws, size_t ws_size,
                              hipStream_t stream) {
  (void)in_sizes; (void)n_in; (void)out_size; (void)ws_size;
  const float* emb   = (const float*)d_in[0];  // (16,128,1024)
  const float* mask  = (const float*)d_in[1];  // (16,128)
  const float* proto = (const float*)d_in[2];  // (1,128,1024,4) == p_flat (128,4096)
  const float* fcw   = (const float*)d_in[3];  // (5,128)
  float* ws  = (float*)d_ws;
  float* out = (float*)d_out;                  // 2048 proto-dists + 80 class-out

  // bf16 arrays carved out of d_ws (all 16B aligned)
  unsigned short* prh = (unsigned short*)(ws + SPLIT_OFF);
  unsigned short* prl = prh + (size_t)PP * KE;
  unsigned short* x0h = prl + (size_t)PP * KE;
  unsigned short* x0l = x0h + (size_t)NN * 125 * KE;
  unsigned short* x1h = x0l + (size_t)NN * 125 * KE;
  unsigned short* x1l = x1h + (size_t)NN * 122 * KE;

  // one-time conversions / gathers
  split_kernel<<<dim3((PP * KE / 4 + 255) / 256), dim3(256), 0, stream>>>(
      proto, prh, prl, PP * KE / 4);
  gather_split_kernel<1, 125><<<dim3((NN * 125 * KE / 4 + 255) / 256),
                                dim3(256), 0, stream>>>(emb, x0h, x0l);
  gather_split_kernel<2, 122><<<dim3((NN * 122 * KE / 4 + 255) / 256),
                                dim3(256), 0, stream>>>(emb, x1h, x1l);

  norms_kernel<<<dim3(2 * NN * SS + PP), dim3(32), 0, stream>>>(emb, proto, ws);
  // branch d=1: H=125, protos [0,64),  overlap=3
  proto_gemm<125, 0, 3, 0><<<dim3(2, 8, NN), dim3(32), 0, stream>>>(
      x0h, x0l, prh, prl, mask, ws);
  // branch d=2: H=122, protos [64,128), overlap=6
  proto_gemm<122, 64, 6, 1><<<dim3(2, 8, NN), dim3(32), 0, stream>>>(
      x1h, x1l, prh, prl, mask, ws);
  finalize_kernel<<<dim3(NN), dim3(PP), 0, stream>>>(fcw, ws, out);
}
